// Encoder3D_53704271069276
// MI455X (gfx1250) — matrix operand order repaired and checked by reading, not compiled, not run on hardware
//
#include <hip/hip_runtime.h>
#include <cstdint>
#include <cstddef>

// ---------------------------------------------------------------------------
// CDNA5 (gfx1250, wave32) implementation of the 3D ViT encoder.
// All matmuls: v_wmma_f32_16x16x32_bf16 (bf16 in, f32 accum).
// Fast kernel: 32x32 tile/wave, branch-free b128 loads (4 wmma / 16 loads).
// Generic kernel: guarded 16x16 tiles for ragged edges (stage 3, T=8).
// ---------------------------------------------------------------------------

typedef __attribute__((ext_vector_type(16))) __bf16 v16bf;
typedef __attribute__((ext_vector_type(8)))  float  v8f;

#define ACT_NONE  0
#define ACT_GELU  1

static __device__ __forceinline__ float gelu_exact(float v) {
  return 0.5f * v * (1.0f + erff(v * 0.70710678118654752f));
}

// unconditional: 8 fp32 -> 8 bf16 (two global_load_b128, 16B-aligned by contract)
static __device__ __forceinline__ void load8f(const float* __restrict__ p, __bf16* dst) {
  float4 a = ((const float4*)p)[0];
  float4 b = ((const float4*)p)[1];
  dst[0] = (__bf16)a.x; dst[1] = (__bf16)a.y; dst[2] = (__bf16)a.z; dst[3] = (__bf16)a.w;
  dst[4] = (__bf16)b.x; dst[5] = (__bf16)b.y; dst[6] = (__bf16)b.z; dst[7] = (__bf16)b.w;
}

// guarded variant for ragged tiles
static __device__ __forceinline__ void load8g(const float* __restrict__ p,
                                              int k0, int Klim, bool ok, __bf16* dst) {
#pragma unroll
  for (int j = 0; j < 8; ++j)
    dst[j] = (ok && (k0 + j) < Klim) ? (__bf16)p[k0 + j] : (__bf16)0.f;
}

// ---------------- fast GEMM: M%32==0, N%32==0, K%32==0, lda/ldb/ldy%4==0 ----
// Y[m,n] = act( alpha * sum_k X[m,k]*Wt[n,k] + bias[n] + sum_r U[m,r]*VL[n,r] ) + beta*Y[m,n]
__global__ void __launch_bounds__(256) wmma_gemm_fast_k(
    const float* __restrict__ X, int lda,
    const float* __restrict__ Wt, int ldb,
    const float* __restrict__ bias,
    const float* __restrict__ U, const float* __restrict__ VL,
    float* __restrict__ Y, int ldy,
    int M, int N, int K, float alpha, float beta, int act) {
  const int lane = threadIdx.x & 31;
  const int wave = threadIdx.x >> 5;
  const int tiles_n = N >> 5;
  const int tile = blockIdx.x * (blockDim.x >> 5) + wave;
  if (tile >= (M >> 5) * tiles_n) return;    // wave-uniform
  const int tm = tile / tiles_n, tn = tile - tm * tiles_n;
  const int half = lane >> 4, r = lane & 15;

  const float* a0p = X  + (size_t)(tm * 32 + r) * lda;
  const float* a1p = a0p + (size_t)16 * lda;
  const float* b0p = Wt + (size_t)(tn * 32 + r) * ldb;
  const float* b1p = b0p + (size_t)16 * ldb;
  // A fragment K offsets (half selects K 0..15 / 16..31)
  const int aoff0 = 16 * half, aoff1 = 16 * half + 8;
  // B fragment K offsets (half selects K 0..15 / 16..31)
  const int boff0 = 16 * half, boff1 = 16 * half + 8;

  v8f c00 = {0.f,0.f,0.f,0.f,0.f,0.f,0.f,0.f};
  v8f c01 = c00, c10 = c00, c11 = c00;

  for (int kk = 0; kk < K; kk += 32) {
    __bf16 a0[16], a1[16], b0[16], b1[16];
    load8f(a0p + kk + aoff0, a0);     load8f(a0p + kk + aoff1, a0 + 8);
    load8f(a1p + kk + aoff0, a1);     load8f(a1p + kk + aoff1, a1 + 8);
    load8f(b0p + kk + boff0, b0);     load8f(b0p + kk + boff1, b0 + 8);
    load8f(b1p + kk + boff0, b1);     load8f(b1p + kk + boff1, b1 + 8);
    if (kk + 32 < K) {
      __builtin_prefetch(a0p + kk + 32, 0, 3);   // -> global_prefetch_b8
      __builtin_prefetch(a1p + kk + 32, 0, 3);
    }
    v16bf av0, av1, bv0, bv1;
#pragma unroll
    for (int j = 0; j < 16; ++j) { av0[j] = a0[j]; av1[j] = a1[j]; bv0[j] = b0[j]; bv1[j] = b1[j]; }
    c00 = __builtin_amdgcn_wmma_f32_16x16x32_bf16(false, av0, false, bv0, (short)0, c00, false, false);
    c01 = __builtin_amdgcn_wmma_f32_16x16x32_bf16(false, av0, false, bv1, (short)0, c01, false, false);
    c10 = __builtin_amdgcn_wmma_f32_16x16x32_bf16(false, av1, false, bv0, (short)0, c10, false, false);
    c11 = __builtin_amdgcn_wmma_f32_16x16x32_bf16(false, av1, false, bv1, (short)0, c11, false, false);
  }

  // epilogue: C layout -> lanes 0-15 vgpr i: M=i; lanes 16-31: M=8+i; N=r
  const int mb = tm * 32 + half * 8;
  const int nb = tn * 32 + r;
  float bs0 = bias ? bias[nb] : 0.f;
  float bs1 = bias ? bias[nb + 16] : 0.f;
  float bl0[8], bl1[8];
  if (VL) {
#pragma unroll
    for (int j = 0; j < 8; ++j) { bl0[j] = VL[(size_t)nb * 8 + j]; bl1[j] = VL[(size_t)(nb + 16) * 8 + j]; }
  }
  const v8f* accs[4] = {&c00, &c01, &c10, &c11};
#pragma unroll
  for (int tq = 0; tq < 4; ++tq) {
    const int mo = mb + (tq >> 1) * 16;
    const int no = (tq & 1) ? nb + 16 : nb;
    const float bsv = (tq & 1) ? bs1 : bs0;
    const float* bl = (tq & 1) ? bl1 : bl0;
#pragma unroll
    for (int i = 0; i < 8; ++i) {
      int m = mo + i;
      float v = alpha * (*accs[tq])[i] + bsv;
      if (U) {
        const float* um = U + (size_t)m * 8;
#pragma unroll
        for (int j = 0; j < 8; ++j) v += um[j] * bl[j];
      }
      if (act == ACT_GELU) v = gelu_exact(v);
      float* yp = Y + (size_t)m * ldy + no;
      if (beta != 0.f) v += beta * (*yp);
      *yp = v;
    }
  }
}

// ---------------- generic guarded GEMM (16x16 tile/wave) --------------------
__global__ void wmma_gemm_gen_k(const float* __restrict__ X, int lda,
                                const float* __restrict__ Wt, int ldb,
                                const float* __restrict__ bias,
                                const float* __restrict__ U, const float* __restrict__ VL,
                                float* __restrict__ Y, int ldy,
                                int M, int N, int K, float alpha, float beta, int act) {
  const int lane = threadIdx.x & 31;
  const int wave = threadIdx.x >> 5;
  const int tiles_m = (M + 15) >> 4;
  const int tiles_n = (N + 15) >> 4;
  const int tile = blockIdx.x * (blockDim.x >> 5) + wave;
  if (tile >= tiles_m * tiles_n) return;
  const int tm = tile / tiles_n, tn = tile - tm * tiles_n;
  const int half = lane >> 4, r = lane & 15;

  const int  arow    = tm * 16 + r;
  const bool arow_ok = arow < M;
  const float* arp   = X + (size_t)arow * lda;
  const int  ncol    = tn * 16 + r;
  const bool ncol_ok = ncol < N;
  const float* brp   = Wt + (size_t)ncol * ldb;

  v8f acc = {0.f,0.f,0.f,0.f,0.f,0.f,0.f,0.f};
  for (int kk = 0; kk < K; kk += 32) {
    __bf16 af[16], bfm[16];
    load8g(arp, kk + 16 * half,     K, arow_ok, af);
    load8g(arp, kk + 16 * half + 8, K, arow_ok, af + 8);
    load8g(brp, kk + 16 * half,     K, ncol_ok, bfm);
    load8g(brp, kk + 16 * half + 8, K, ncol_ok, bfm + 8);
    v16bf av, bv;
#pragma unroll
    for (int j = 0; j < 16; ++j) { av[j] = af[j]; bv[j] = bfm[j]; }
    acc = __builtin_amdgcn_wmma_f32_16x16x32_bf16(false, av, false, bv, (short)0, acc, false, false);
  }
  if (!ncol_ok) return;
  float bsv = bias ? bias[ncol] : 0.f;
  float bl[8];
  if (VL) {
#pragma unroll
    for (int j = 0; j < 8; ++j) bl[j] = VL[(size_t)ncol * 8 + j];
  }
#pragma unroll
  for (int i = 0; i < 8; ++i) {
    int m = tm * 16 + half * 8 + i;
    if (m >= M) continue;
    float v = alpha * acc[i] + bsv;
    if (U) {
      const float* um = U + (size_t)m * 8;
#pragma unroll
      for (int j = 0; j < 8; ++j) v += um[j] * bl[j];
    }
    if (act == ACT_GELU) v = gelu_exact(v);
    float* yp = Y + (size_t)m * ldy + ncol;
    if (beta != 0.f) v += beta * (*yp);
    *yp = v;
  }
}

// -------------------------- support kernels --------------------------------

__global__ void layer_norm_k(const float* __restrict__ X, const float* __restrict__ g,
                             const float* __restrict__ b, float* __restrict__ Y, int C) {
  __shared__ float sbuf[256];
  const int row = blockIdx.x, tid = threadIdx.x;
  const float* xr = X + (size_t)row * C;
  float s = 0.f;
  for (int c = tid; c < C; c += 256) s += xr[c];
  sbuf[tid] = s; __syncthreads();
  for (int st = 128; st > 0; st >>= 1) { if (tid < st) sbuf[tid] += sbuf[tid + st]; __syncthreads(); }
  const float mu = sbuf[0] / (float)C;
  __syncthreads();
  float v = 0.f;
  for (int c = tid; c < C; c += 256) { float d = xr[c] - mu; v += d * d; }
  sbuf[tid] = v; __syncthreads();
  for (int st = 128; st > 0; st >>= 1) { if (tid < st) sbuf[tid] += sbuf[tid + st]; __syncthreads(); }
  const float rstd = rsqrtf(sbuf[0] / (float)C + 1e-5f);
  float* yr = Y + (size_t)row * C;
  for (int c = tid; c < C; c += 256) yr[c] = (xr[c] - mu) * rstd * g[c] + b[c];
}

__global__ void softmax_k(float* __restrict__ S, int L, int ld) {
  __shared__ float sbuf[256];
  const int row = blockIdx.x, tid = threadIdx.x;
  float* sr = S + (size_t)row * ld;
  float mx = -3.4e38f;
  for (int c = tid; c < L; c += 256) mx = fmaxf(mx, sr[c]);
  sbuf[tid] = mx; __syncthreads();
  for (int st = 128; st > 0; st >>= 1) { if (tid < st) sbuf[tid] = fmaxf(sbuf[tid], sbuf[tid + st]); __syncthreads(); }
  mx = sbuf[0]; __syncthreads();
  float sum = 0.f;
  for (int c = tid; c < L; c += 256) { float e = expf(sr[c] - mx); sr[c] = e; sum += e; }
  sbuf[tid] = sum; __syncthreads();
  for (int st = 128; st > 0; st >>= 1) { if (tid < st) sbuf[tid] += sbuf[tid + st]; __syncthreads(); }
  const float inv = 1.0f / sbuf[0];
  for (int c = tid; c < L; c += 256) sr[c] *= inv;
}

// u[m,r] = sum_k X[m,k] * A[r,k]
__global__ void lora_u_k(const float* __restrict__ X, const float* __restrict__ A,
                         float* __restrict__ U, int Mtok, int C) {
  int i = blockIdx.x * blockDim.x + threadIdx.x;
  if (i >= Mtok * 8) return;
  int m = i >> 3, r = i & 7;
  const float* xr = X + (size_t)m * C;
  const float* ar = A + (size_t)r * C;
  float s = 0.f;
  for (int c = 0; c < C; ++c) s += xr[c] * ar[c];
  U[i] = s;
}

// top-2-of-4 gating, softmax over selected logits (ties -> lower index)
__global__ void gate_topk_k(const float* __restrict__ X, const float* __restrict__ gw,
                            const float* __restrict__ gb, float* __restrict__ wout,
                            int* __restrict__ iout, int Ntok, int C) {
  int m = blockIdx.x * blockDim.x + threadIdx.x;
  if (m >= Ntok) return;
  const float* xr = X + (size_t)m * C;
  float lg[4];
  for (int e = 0; e < 4; ++e) {
    float s = gb[e];
    const float* w = gw + (size_t)e * C;
    for (int c = 0; c < C; ++c) s += xr[c] * w[c];
    lg[e] = s;
  }
  int i0 = 0;
  for (int e = 1; e < 4; ++e) if (lg[e] > lg[i0]) i0 = e;
  int i1 = -1;
  for (int e = 0; e < 4; ++e) { if (e == i0) continue; if (i1 < 0 || lg[e] > lg[i1]) i1 = e; }
  float e1 = expf(lg[i1] - lg[i0]);
  wout[m * 2] = 1.0f / (1.0f + e1);
  wout[m * 2 + 1] = e1 / (1.0f + e1);
  iout[m * 2] = i0; iout[m * 2 + 1] = i1;
}

// t[m,c] += sum_k w[m,k] * y[idx[m,k], m, c]
__global__ void moe_combine_k(float* __restrict__ t, const float* __restrict__ ya,
                              const float* __restrict__ w2, const int* __restrict__ i2,
                              int Ntok, int C) {
  int i = blockIdx.x * blockDim.x + threadIdx.x;
  if (i >= Ntok * C) return;
  int m = i / C, c = i - m * C;
  float acc = t[i];
  acc += w2[m * 2]     * ya[((size_t)i2[m * 2]     * Ntok + m) * C + c];
  acc += w2[m * 2 + 1] * ya[((size_t)i2[m * 2 + 1] * Ntok + m) * C + c];
  t[i] = acc;
}

// im2col for the 4^3/stride-4 patch conv, one input group (64^3 volume)
__global__ void im2col_patch_k(const float* __restrict__ x, float* __restrict__ xc, int g) {
  int i = blockIdx.x * blockDim.x + threadIdx.x;
  if (i >= 4096 * 64) return;
  int tok = i >> 6, k = i & 63;
  int od = tok >> 8, oh = (tok >> 4) & 15, ow = tok & 15;
  int kd = k >> 4,  kh = (k >> 2) & 3,    kw = k & 3;
  xc[i] = x[(size_t)g * 262144 + (size_t)(od * 4 + kd) * 4096 + (oh * 4 + kh) * 64 + (ow * 4 + kw)];
}

// im2col for 2^3/stride-2 down-projection convs; input tokens [Din^3, C]
__global__ void im2col_proj_k(const float* __restrict__ tin, float* __restrict__ xc,
                              int Din, int C) {
  const int Dout = Din >> 1;
  const int Tout = Dout * Dout * Dout;
  const int K = 8 * C;
  int i = blockIdx.x * blockDim.x + threadIdx.x;
  if (i >= Tout * K) return;
  int tok = i / K, k = i - tok * K;
  int ic = k >> 3, kk = k & 7;
  int kd = kk >> 2, kh = (kk >> 1) & 1, kw = kk & 1;
  int od = tok / (Dout * Dout), rem = tok - od * (Dout * Dout);
  int oh = rem / Dout, ow = rem - oh * Dout;
  int tin_idx = (2 * od + kd) * Din * Din + (2 * oh + kh) * Din + (2 * ow + kw);
  xc[i] = tin[(size_t)tin_idx * C + ic];
}

// [T,C] token-major -> [C,T] channel-major (also used for the stage outputs)
__global__ void transpose_k(const float* __restrict__ t, float* __restrict__ out,
                            int Ttok, int C) {
  int i = blockIdx.x * blockDim.x + threadIdx.x;
  if (i >= Ttok * C) return;
  int tok = i / C, c = i - tok * C;
  out[(size_t)c * Ttok + tok] = t[i];
}

// ------------------------------- host side ---------------------------------

static inline int cdiv(int a, int b) { return (a + b - 1) / b; }

static void gemm(hipStream_t st, const float* X, int lda, const float* Wt, int ldb,
                 const float* bias, const float* U, const float* VL,
                 float* Y, int ldy, int M, int N, int K, float alpha, float beta, int act) {
  bool fast = (M % 32 == 0) && (N % 32 == 0) && (K % 32 == 0) &&
              (lda % 4 == 0) && (ldb % 4 == 0) && (ldy % 4 == 0);
  if (fast) {
    int tiles = (M >> 5) * (N >> 5);
    wmma_gemm_fast_k<<<cdiv(tiles, 8), 256, 0, st>>>(X, lda, Wt, ldb, bias, U, VL,
                                                     Y, ldy, M, N, K, alpha, beta, act);
  } else {
    int tiles = cdiv(M, 16) * cdiv(N, 16);
    wmma_gemm_gen_k<<<cdiv(tiles, 8), 256, 0, st>>>(X, lda, Wt, ldb, bias, U, VL,
                                                    Y, ldy, M, N, K, alpha, beta, act);
  }
}

struct LoraP  { const float *w, *b, *A, *B; };
struct BlockP { LoraP q, k, v, o;
                const float *ln1_g, *ln1_b, *ln2_g, *ln2_b;
                const float *gate_w, *gate_b, *w1, *b1, *w2, *b2; };
struct ProjP  { const float *w, *b; };

extern "C" void kernel_launch(void* const* d_in, const int* in_sizes, int n_in,
                              void* d_out, int out_size, void* d_ws, size_t ws_size,
                              hipStream_t stream) {
  (void)out_size; (void)ws_size;
  // ---- locate x and params (handles sorted-pytree vs insertion-order flattening)
  const float* x;
  int base;
  if (in_sizes[0] == 1048576) { x = (const float*)d_in[0]; base = 1; }
  else                        { x = (const float*)d_in[n_in - 1]; base = 0; }
  bool sorted_order = (in_sizes[base] == 2048);  // blocks[0].attn.k.A is [8,256]

  int pi = base;
  auto nxt = [&]() { return (const float*)d_in[pi++]; };

  BlockP blk[5]; ProjP proj[3];
  const float *patch_w, *patch_b;
  if (sorted_order) {
    for (int bi = 0; bi < 5; ++bi) {
      BlockP& B = blk[bi];
      LoraP* ls[4] = {&B.k, &B.o, &B.q, &B.v};               // attn keys sorted: k,o,q,v
      for (int j = 0; j < 4; ++j) { ls[j]->A = nxt(); ls[j]->B = nxt(); ls[j]->b = nxt(); ls[j]->w = nxt(); }
      B.b1 = nxt(); B.b2 = nxt(); B.gate_b = nxt(); B.gate_w = nxt();
      B.ln1_b = nxt(); B.ln1_g = nxt(); B.ln2_b = nxt(); B.ln2_g = nxt();
      B.w1 = nxt(); B.w2 = nxt();
    }
    patch_b = nxt(); patch_w = nxt();
    for (int i = 0; i < 3; ++i) { proj[i].b = nxt(); proj[i].w = nxt(); }
  } else {
    patch_w = nxt(); patch_b = nxt();
    for (int i = 0; i < 3; ++i) { proj[i].w = nxt(); proj[i].b = nxt(); }
    for (int bi = 0; bi < 5; ++bi) {
      BlockP& B = blk[bi];
      B.ln1_g = nxt(); B.ln1_b = nxt(); B.ln2_g = nxt(); B.ln2_b = nxt();
      LoraP* ls[4] = {&B.q, &B.k, &B.v, &B.o};
      for (int j = 0; j < 4; ++j) { ls[j]->w = nxt(); ls[j]->b = nxt(); ls[j]->A = nxt(); ls[j]->B = nxt(); }
      B.gate_w = nxt(); B.gate_b = nxt(); B.w1 = nxt(); B.b1 = nxt(); B.w2 = nxt(); B.b2 = nxt();
    }
  }

  // ---- workspace carve-up (floats); total ~21.0M floats ~= 84 MB
  float* W  = (float*)d_ws;
  float* t  = W;                 // [4096,256] max
  float* tl = W + 1048576;
  float* qb = W + 2097152;
  float* kb = W + 3145728;
  float* vb = W + 4194304;
  float* ao = W + 5242880;
  float* ub = W + 6291456;       // [T,8]
  float* sc = W + 6324224;       // [1024, 4096] score chunk
  float* hb = W + 10518528;      // [T, 4C] MoE hidden
  float* ya = W + 14712832;      // [4, T, C] expert outputs
  float* xc = W + 18907136;      // im2col, max 1,048,576
  float* vt = W + 19955712;      // V transposed [C, T], max 1,048,576
  float* g2 = W + 21004288;      // gate weights [T,2]
  int*   i2 = (int*)(W + 21012480);

  const int DIMSs[4]  = {256, 512, 1024, 2048};
  const int DEPTHs[4] = {1, 1, 2, 1};
  const int HEADSs[4] = {2, 4, 4, 8};
  const int DSP[4]    = {16, 8, 4, 2};
  const int TTOK[4]   = {4096, 512, 64, 8};
  const int outOff[3] = {0, 1048576, 1310720};

  // ---- patch embed: 4 groups of im2col + WMMA gemm, concatenated on channels
  for (int g = 0; g < 4; ++g) {
    im2col_patch_k<<<cdiv(4096 * 64, 256), 256, 0, stream>>>(x, xc, g);
    gemm(stream, xc, 64, patch_w + (size_t)g * 4096, 64, patch_b + g * 64,
         nullptr, nullptr, t + g * 64, 256, 4096, 64, 64, 1.f, 0.f, ACT_NONE);
  }

  int bi = 0;
  for (int s = 0; s < 4; ++s) {
    const int C = DIMSs[s], T = TTOK[s], h = HEADSs[s], hc = C / h, hd = 4 * C;
    for (int dd = 0; dd < DEPTHs[s]; ++dd, ++bi) {
      const BlockP& B = blk[bi];
      // ---- attention: t = t + O(attn(LN1(t)))
      layer_norm_k<<<T, 256, 0, stream>>>(t, B.ln1_g, B.ln1_b, tl, C);
      const LoraP* lps[3] = {&B.q, &B.k, &B.v};
      float* outs3[3] = {qb, kb, vb};
      for (int j = 0; j < 3; ++j) {
        lora_u_k<<<cdiv(T * 8, 256), 256, 0, stream>>>(tl, lps[j]->A, ub, T, C);
        gemm(stream, tl, C, lps[j]->w, C, lps[j]->b, ub, lps[j]->B,
             outs3[j], C, T, C, C, 1.f, 0.f, ACT_NONE);
      }
      // transpose V -> [C, T] so attn@V uses contiguous-K WMMA loads
      transpose_k<<<cdiv(T * C, 256), 256, 0, stream>>>(vb, vt, T, C);
      const float scale = rsqrtf((float)hc);
      for (int hh = 0; hh < h; ++hh) {
        for (int q0 = 0; q0 < T; q0 += 1024) {
          int qm = (T - q0 < 1024) ? (T - q0) : 1024;
          // scores = scale * Q_h K_h^T  (Wt role: K rows, ldb=C)
          gemm(stream, qb + (size_t)q0 * C + hh * hc, C, kb + hh * hc, C,
               nullptr, nullptr, nullptr, sc, T, qm, T, hc, scale, 0.f, ACT_NONE);
          softmax_k<<<qm, 256, 0, stream>>>(sc, T, T);
          // out_h = attn @ V_h  (Wt role: vt rows = channels, ldb=T)
          gemm(stream, sc, T, vt + (size_t)hh * hc * T, T,
               nullptr, nullptr, nullptr, ao + (size_t)q0 * C + hh * hc, C,
               qm, hc, T, 1.f, 0.f, ACT_NONE);
        }
      }
      lora_u_k<<<cdiv(T * 8, 256), 256, 0, stream>>>(ao, B.o.A, ub, T, C);
      gemm(stream, ao, C, B.o.w, C, B.o.b, ub, B.o.B,
           t, C, T, C, C, 1.f, 1.f, ACT_NONE);     // beta=1 -> residual add
      // ---- MoE: t = t + combine(top2, experts(LN2(t)))
      layer_norm_k<<<T, 256, 0, stream>>>(t, B.ln2_g, B.ln2_b, tl, C);
      gate_topk_k<<<cdiv(T, 128), 128, 0, stream>>>(tl, B.gate_w, B.gate_b, g2, i2, T, C);
      for (int e = 0; e < 4; ++e) {
        gemm(stream, tl, C, B.w1 + (size_t)e * hd * C, C, B.b1 + e * hd,
             nullptr, nullptr, hb, hd, T, hd, C, 1.f, 0.f, ACT_GELU);
        gemm(stream, hb, hd, B.w2 + (size_t)e * C * hd, hd, B.b2 + e * C,
             nullptr, nullptr, ya + (size_t)e * T * C, C, T, C, hd, 1.f, 0.f, ACT_NONE);
      }
      moe_combine_k<<<cdiv(T * C, 256), 256, 0, stream>>>(t, ya, g2, i2, T, C);
    }
    if (s < 3) {
      // emit stage feature map [C, D, H, W], then down-project to next stage
      transpose_k<<<cdiv(T * C, 256), 256, 0, stream>>>(t, (float*)d_out + outOff[s], T, C);
      const int Tn = TTOK[s + 1], Cn = DIMSs[s + 1], Kp = 8 * C;
      im2col_proj_k<<<cdiv(Tn * Kp, 256), 256, 0, stream>>>(t, xc, DSP[s], C);
      gemm(stream, xc, Kp, proj[s].w, Kp, proj[s].b, nullptr, nullptr,
           t, Cn, Tn, Cn, Kp, 1.f, 0.f, ACT_NONE);
    }
  }
}
